// PredictorPerCT_19146964206238
// MI455X (gfx1250) — matrix operand order repaired and checked by reading, not compiled, not run on hardware
//
#include <hip/hip_runtime.h>

typedef __attribute__((ext_vector_type(16))) __bf16 v16bf;
typedef __attribute__((ext_vector_type(8)))  float  v8f;

constexpr int N    = 16384;
constexpr int D    = 1024;
constexpr int E    = 8;
constexpr int DOUT = 1024;

constexpr int ROWS_PER_TILE  = 32;    // two 16-row WMMA subtiles per block
constexpr int COLS_PER_BLOCK = 256;   // 8 waves x 2 column-groups (16 cols each)
constexpr int KSTEP          = 32;    // bf16 WMMA K
constexpr int KSLAB          = 64;    // K staged per async slab (2 WMMA K-steps)
constexpr int MAX_TILES      = N / ROWS_PER_TILE + E;   // 520 upper bound
constexpr int ASTRIDE        = 72;    // padded sA row stride (ushorts) -> conflict-free b128 frag reads

__device__ __forceinline__ unsigned short f2bf(float f) {
    unsigned u = __float_as_uint(f);
    u += 0x7FFFu + ((u >> 16) & 1u);          // round-to-nearest-even
    return (unsigned short)(u >> 16);
}

__device__ __forceinline__ int argmax8(const float* __restrict__ t) {
    int best = 0; float bv = t[0];
#pragma unroll
    for (int j = 1; j < E; ++j) {
        float v = t[j];
        if (v > bv) { bv = v; best = j; }     // strict > keeps first occurrence
    }
    return best;
}

// Async global->LDS copy of 16 bytes per lane (GVS mode: SGPR64 base + VGPR32 offset).
// Tracked by ASYNCcnt; completion enforced with s_wait_asynccnt.
__device__ __forceinline__ void async_copy_b128(unsigned lds_addr, unsigned voff,
                                                unsigned long long sbase) {
    asm volatile("global_load_async_to_lds_b128 %0, %1, %2"
                 :: "v"(lds_addr), "v"(voff), "s"(sbase) : "memory");
}
__device__ __forceinline__ void wait_async0() {
    asm volatile("s_wait_asynccnt 0x0" ::: "memory");
}

// ---------------- one-time bf16 pre-conversion ----------------

// x (f32, [N][D]) -> xb (bf16, [N][D] row-major)
__global__ void cvt_x(const float* __restrict__ x, unsigned short* __restrict__ xb) {
    size_t i = ((size_t)blockIdx.x * blockDim.x + threadIdx.x) * 4;
    float4 v = *(const float4*)(x + i);
    uint2 o;
    o.x = (unsigned)f2bf(v.x) | ((unsigned)f2bf(v.y) << 16);
    o.y = (unsigned)f2bf(v.z) | ((unsigned)f2bf(v.w) << 16);
    *(uint2*)(xb + i) = o;
}

// W (f32, [E][D][DOUT]) -> Wb (bf16) pre-tiled in WMMA B-fragment order:
//   Wb[ e*D*DOUT + (c/16)*(D*16) + (k/16)*256 + (c%16)*16 + (k%16) ]
// Column-major inside each 16(K)x16(N) tile, so a lane's fragment half
// (contiguous K for fixed column, the k order this kernel uses for A and B) is 16 contiguous bf16.
__global__ void cvt_W(const float* __restrict__ W, unsigned short* __restrict__ Wb) {
    size_t t  = (size_t)blockIdx.x * blockDim.x + threadIdx.x; // E*(D/4)*DOUT threads
    int c  = (int)(t % DOUT);
    int kq = (int)((t / DOUT) % (D / 4));          // handles k = 4*kq .. 4*kq+3
    int e  = (int)(t / ((size_t)DOUT * (D / 4)));
    const float* src = W + ((size_t)e * D + 4 * kq) * DOUT + c;
    unsigned short r0 = f2bf(src[0]);
    unsigned short r1 = f2bf(src[DOUT]);
    unsigned short r2 = f2bf(src[2 * DOUT]);
    unsigned short r3 = f2bf(src[3 * DOUT]);
    size_t off = (size_t)e * D * DOUT + (size_t)(c >> 4) * (D * 16)
               + (size_t)(kq >> 2) * 256 + (size_t)(c & 15) * 16 + (kq & 3) * 4;
    uint2 o;
    o.x = (unsigned)r0 | ((unsigned)r1 << 16);
    o.y = (unsigned)r2 | ((unsigned)r3 << 16);
    *(uint2*)(Wb + off) = o;
}

// ---------------- routing kernels ----------------

__global__ void moe_count(const float* __restrict__ ten, int* __restrict__ counts) {
    int n = blockIdx.x * blockDim.x + threadIdx.x;
    if (n >= N) return;
    atomicAdd(&counts[argmax8(ten + (size_t)n * E)], 1);
}

__global__ void moe_scan(const int* __restrict__ counts,
                         int* __restrict__ tileStart,
                         int* __restrict__ rowStart,
                         int* __restrict__ cursor) {
    if (threadIdx.x == 0 && blockIdx.x == 0) {
        int tAcc = 0, rAcc = 0;
        for (int e = 0; e < E; ++e) {
            tileStart[e] = tAcc;
            rowStart[e]  = rAcc;
            cursor[e]    = 0;
            tAcc += (counts[e] + ROWS_PER_TILE - 1) / ROWS_PER_TILE;
            rAcc += counts[e];
        }
        tileStart[E] = tAcc;
        rowStart[E]  = rAcc;
    }
}

__global__ void moe_place(const float* __restrict__ ten,
                          const int* __restrict__ rowStart,
                          int* __restrict__ cursor,
                          int* __restrict__ perm) {
    int n = blockIdx.x * blockDim.x + threadIdx.x;
    if (n >= N) return;
    int e = argmax8(ten + (size_t)n * E);
    int pos = atomicAdd(&cursor[e], 1);
    perm[rowStart[e] + pos] = n;
}

// ---------------- gathered WMMA GEMM (bf16 in, f32 out) ----------------

union FragB16 { v16bf v; uint4 q[2]; };

__global__ __launch_bounds__(256)
void moe_gemm(const unsigned short* __restrict__ xb,
              const unsigned short* __restrict__ Wb,
              const float* __restrict__ bias,
              const int*   __restrict__ perm,
              const int*   __restrict__ tileStart,
              const int*   __restrict__ counts,
              const int*   __restrict__ rowStart,
              float*       __restrict__ out) {
    __shared__ int            sRows[ROWS_PER_TILE];
    __shared__ unsigned short sA[2][ROWS_PER_TILE * ASTRIDE];   // double-buffered 32x64 A slab

    const int t = blockIdx.x;
    if (t >= tileStart[E]) return;                 // block-uniform exit

    int e = 0;
#pragma unroll
    for (int q = E - 1; q > 0; --q) {
        if (t >= tileStart[q]) { e = q; break; }
    }
    const int tloc  = t - tileStart[e];
    const int cnt   = counts[e];
    const int rbase = rowStart[e] + tloc * ROWS_PER_TILE;

    const int tid = threadIdx.x;
    if (tid < ROWS_PER_TILE) {
        int idx = tloc * ROWS_PER_TILE + tid;
        sRows[tid] = (idx < cnt) ? perm[rbase + tid] : -1;
    }
    __syncthreads();

    const int lane = tid & 31;
    const int wave = tid >> 5;          // 0..7
    const int half = lane >> 4;         // lane half, 0/1
    const int lm   = lane & 15;

    // ---- A staging: 32 rows x 64 k bf16 slab = 4KB, 8 ushorts (16B) per thread ----
    const int la = tid * 8;
    const int ar = la >> 6;             // row in tile
    const int ak = la & 63;             // k offset (multiple of 8)
    const int rowA = sRows[ar];
    const unsigned lds0 = (unsigned)(size_t)&sA[0][ar * ASTRIDE + ak];
    const unsigned lds1 = (unsigned)(size_t)&sA[1][ar * ASTRIDE + ak];
    const unsigned voff0 = (unsigned)(((size_t)(rowA >= 0 ? rowA : 0) * D + ak) * 2);
    const unsigned long long xbase = (unsigned long long)xb;

    // ---- B: direct fragment loads from pre-tiled Wb, two column groups per wave ----
    const int cg0 = blockIdx.y * (COLS_PER_BLOCK / 16) + wave * 2;   // 0..63
    const unsigned short* wb = Wb + (size_t)e * D * DOUT + (size_t)cg0 * (D * 16)
                             + half * 256 + lm * 16;
    constexpr int CGOFF = D * 16;                   // ushorts between column groups

    // ---- A fragment LDS base offsets (two 16B chunks per fragment per K-step) ----
    const int a0 = lm * ASTRIDE + 16 * half;
    const int a1 = (lm + 16) * ASTRIDE + 16 * half;

    v8f acc00 = {}, acc10 = {}, acc01 = {}, acc11 = {};

    // prologue: kick off first A slab
    async_copy_b128(lds0, voff0, xbase);

    for (int ks = 0; ks < D; ks += KSLAB) {
        const int cur = (ks >> 6) & 1;
        wait_async0();                   // own slab copy done
        __syncthreads();                 // whole block's slab visible; prev readers done

        if (ks + KSLAB < D)              // overlap next slab copy with this stage's math
            async_copy_b128(cur ? lds0 : lds1, voff0 + (ks + KSLAB) * 2, xbase);

        const unsigned short* aBuf = sA[cur];
#pragma unroll
        for (int kk = 0; kk < KSLAB; kk += KSTEP) {
            FragB16 fa0, fa1, fb0, fb1;
            fa0.q[0] = *(const uint4*)(aBuf + a0 + kk);
            fa0.q[1] = *(const uint4*)(aBuf + a0 + kk + 8);
            fa1.q[0] = *(const uint4*)(aBuf + a1 + kk);
            fa1.q[1] = *(const uint4*)(aBuf + a1 + kk + 8);

            const int k0 = ks + kk;
            const uint4* bp  = (const uint4*)(wb + (size_t)(k0 >> 4) * 256);
            const uint4* bp1 = (const uint4*)(wb + CGOFF + (size_t)(k0 >> 4) * 256);
            fb0.q[0] = bp[0];
            fb0.q[1] = bp[1];
            fb1.q[0] = bp1[0];
            fb1.q[1] = bp1[1];
            __builtin_prefetch(bp + 32, 0, 1);    // next K-step tiles for cg0
            __builtin_prefetch(bp1 + 32, 0, 1);   // next K-step tiles for cg1

            acc00 = __builtin_amdgcn_wmma_f32_16x16x32_bf16(
                        false, fa0.v, false, fb0.v, (short)0, acc00, false, false);
            acc10 = __builtin_amdgcn_wmma_f32_16x16x32_bf16(
                        false, fa1.v, false, fb0.v, (short)0, acc10, false, false);
            acc01 = __builtin_amdgcn_wmma_f32_16x16x32_bf16(
                        false, fa0.v, false, fb1.v, (short)0, acc01, false, false);
            acc11 = __builtin_amdgcn_wmma_f32_16x16x32_bf16(
                        false, fa1.v, false, fb1.v, (short)0, acc11, false, false);
        }
    }

    // ---- epilogue: bias + scatter via perm ----
    const int col0 = blockIdx.y * COLS_PER_BLOCK + wave * 32 + lm;   // cg0 column
    const int col1 = col0 + 16;                                      // cg1 column
    const float bv0 = bias[(size_t)e * DOUT + col0];
    const float bv1 = bias[(size_t)e * DOUT + col1];
#pragma unroll
    for (int i = 0; i < 8; ++i) {
        int m = i + 8 * half;                 // C layout: VGPR i -> M = i + 8*laneHalf
        int r0 = sRows[m];
        if (r0 >= 0) {
            out[(size_t)r0 * DOUT + col0] = acc00[i] + bv0;
            out[(size_t)r0 * DOUT + col1] = acc01[i] + bv1;
        }
        int r1 = sRows[m + 16];
        if (r1 >= 0) {
            out[(size_t)r1 * DOUT + col0] = acc10[i] + bv0;
            out[(size_t)r1 * DOUT + col1] = acc11[i] + bv1;
        }
    }
}

// ---------------- launcher ----------------

extern "C" void kernel_launch(void* const* d_in, const int* in_sizes, int n_in,
                              void* d_out, int out_size, void* d_ws, size_t ws_size,
                              hipStream_t stream) {
    const float* x   = (const float*)d_in[0];   // [N, D]
    const float* ten = (const float*)d_in[1];   // [N, E]
    const float* W   = (const float*)d_in[2];   // [E, D, DOUT]
    const float* b   = (const float*)d_in[3];   // [E, DOUT]
    float* out = (float*)d_out;                 // [N, DOUT]

    char* ws = (char*)d_ws;
    int* counts    = (int*)(ws + 0);            // E
    int* cursor    = (int*)(ws + 64);           // E
    int* tileStart = (int*)(ws + 128);          // E+1
    int* rowStart  = (int*)(ws + 256);          // E+1
    int* perm      = (int*)(ws + 512);          // N ints (64KB)
    unsigned short* xb = (unsigned short*)(ws + (1 << 17));                      // 32MB bf16 x
    unsigned short* Wb = (unsigned short*)(ws + (1 << 17) + (size_t)N * D * 2);  // 16MB bf16 W (tiled)

    hipMemsetAsync(counts, 0, E * sizeof(int), stream);
    moe_count<<<N / 256, 256, 0, stream>>>(ten, counts);
    moe_scan<<<1, 32, 0, stream>>>(counts, tileStart, rowStart, cursor);
    moe_place<<<N / 256, 256, 0, stream>>>(ten, rowStart, cursor, perm);

    cvt_x<<<(N * (size_t)D) / (4 * 256), 256, 0, stream>>>(x, xb);
    cvt_W<<<((size_t)E * D * DOUT) / (4 * 256), 256, 0, stream>>>(W, Wb);

    dim3 grid(MAX_TILES, DOUT / COLS_PER_BLOCK);
    moe_gemm<<<grid, 256, 0, stream>>>(xb, Wb, b, perm, tileStart, counts, rowStart, out);
}
